// CombineLSTM_83176336654932
// MI455X (gfx1250) — compile-verified, matrix operand order checked by reading, not run on hardware
//
#include <hip/hip_runtime.h>
#include <hip/hip_bf16.h>

typedef __attribute__((ext_vector_type(16))) _Float16 v16h;
typedef __attribute__((ext_vector_type(8)))  float    v8f;

#define WMMA_F16(a, b, c) \
  __builtin_amdgcn_wmma_f32_16x16x32_f16(false, (a), false, (b), (short)0, (c), false, false)

__device__ __forceinline__ v8f vzero8() {
  v8f v;
#pragma unroll
  for (int i = 0; i < 8; ++i) v[i] = 0.0f;
  return v;
}

// Load a 16x32 f16 fragment (A layout; B uses the same per-lane pattern with
// rows = N columns). Row-major source, ld in elements. Lane l: row row0+(l&15),
// K segments at k0+kg*8 and k0+16+kg*8, kg=l>>4. 16B-aligned when ld%8==0, k0%8==0.
__device__ __forceinline__ v16h load_frag(const _Float16* p, int ld, int row0, int k0) {
  int lane = threadIdx.x & 31;
  int r = lane & 15, kg = lane >> 4;
  const _Float16* b0 = p + (size_t)(row0 + r) * ld + (k0 + kg * 8);
  v16h v;
  ((uint4*)&v)[0] = *(const uint4*)(b0);
  ((uint4*)&v)[1] = *(const uint4*)(b0 + 16);
  return v;
}

__device__ __forceinline__ float sigm(float x)  { return 1.0f / (1.0f + __expf(-x)); }
__device__ __forceinline__ float tanhf_(float x){ return 2.0f * sigm(2.0f * x) - 1.0f; }

// ---------------- weight conversion / padding ----------------
__global__ void cvt_pad_f16(const float* __restrict__ src, _Float16* __restrict__ dst,
                            int R, int C, int Rp, int Cp) {
  int gid = blockIdx.x * blockDim.x + threadIdx.x;
  if (gid >= Rp * Cp) return;
  int r = gid / Cp, c = gid % Cp;
  float v = (r < R && c < C) ? src[(size_t)r * C + c] : 0.0f;
  dst[gid] = (_Float16)v;
}

__global__ void pad_bias(const float* __restrict__ src, float* __restrict__ dst, int N, int Np) {
  int gid = blockIdx.x * blockDim.x + threadIdx.x;
  if (gid >= Np) return;
  dst[gid] = (gid < N) ? src[gid] : 0.0f;
}

// ---------------- embedding gather (concat w2v|glove), f16, cols padded ----------------
__global__ void embed_gather(const int* __restrict__ wids, const int* __restrict__ gids,
                             const float* __restrict__ w2v, const float* __restrict__ glove,
                             _Float16* __restrict__ out, int rows, int Cp) {
  int gid = blockIdx.x * blockDim.x + threadIdx.x;
  if (gid >= rows * Cp) return;
  int row = gid / Cp, c = gid % Cp;           // row = b*256 + l
  float v = 0.0f;
  if (c < 300)       v = w2v[(size_t)wids[row] * 300 + c];
  else if (c < 400)  v = glove[(size_t)gids[row] * 100 + (c - 300)];
  out[gid] = (_Float16)v;
}

// ---------------- WMMA GEMM: C[M,N] = A[M,K] * B[N,K]^T + bias ----------------
// Block = 8 waves stacked along M (block tile 128 x 64); each wave owns a
// 16(M) x 64(N) strip = 4 N-tiles, all unconditionally valid:
// requires M%128==0, N%64==0, K%32==0.  Grid: (M/128, N/64).
template <bool F32OUT, bool F16OUT>
__global__ __launch_bounds__(256) void gemm_nt(const _Float16* __restrict__ A, int lda,
                                               const _Float16* __restrict__ B, int ldb,
                                               const float* __restrict__ bias,
                                               float* __restrict__ Cf, _Float16* __restrict__ Ch,
                                               int ldc, int K) {
  int wave = threadIdx.x >> 5;
  int m0 = blockIdx.x * 128 + wave * 16;
  int n0 = blockIdx.y * 64;

  v8f acc[4];
#pragma unroll
  for (int i = 0; i < 4; ++i) acc[i] = vzero8();

  for (int kt = 0; kt < K; kt += 32) {
    if (kt + 32 < K) {   // speculative prefetch of next K-slice (global_prefetch_b8)
      __builtin_prefetch(A + (size_t)m0 * lda + kt + 32, 0, 1);
      __builtin_prefetch(B + (size_t)n0 * ldb + kt + 32, 0, 1);
    }
    v16h a = load_frag(A, lda, m0, kt);
#pragma unroll
    for (int i = 0; i < 4; ++i) {
      v16h b = load_frag(B, ldb, n0 + i * 16, kt);
      acc[i] = WMMA_F16(a, b, acc[i]);
    }
  }

  int lane = threadIdx.x & 31, kg = lane >> 4, nn = lane & 15;
#pragma unroll
  for (int i = 0; i < 4; ++i) {
    int n = n0 + i * 16 + nn;
    float bv = bias[n];
#pragma unroll
    for (int r = 0; r < 8; ++r) {
      float v = acc[i][r] + bv;
      size_t off = (size_t)(m0 + r + 8 * kg) * ldc + n;
      if (F32OUT) Cf[off] = v;
      if (F16OUT) Ch[off] = (_Float16)v;
    }
  }
}

// ---------------- bidirectional LSTM scan ----------------
// Grid = 2 (dir 0 fwd, dir 1 bwd), 512 threads = 16 waves.
// Wave w owns hidden units [16w,16w+16): its gate tiles sit at n0 = g*256+16w,
// so i/f/g/o for those units never leave the wave. h lives in LDS (16 KB),
// cell state lives in registers.
// W_hh handling: K-tiles 0..2 (96 VGPRs of B fragments) are register-resident
// across all 256 steps; K-tiles 3..7 are streamed from L2 each step. The
// asm memory barrier keeps the streamed loads inside the step loop (prevents
// LICM hoist -> scratch spill), and the rolled sub-loop bounds live VGPRs.
__global__ __launch_bounds__(512) void lstm_scan(const float* __restrict__ xpf,
                                                 const float* __restrict__ xpb,
                                                 const _Float16* __restrict__ Wf,
                                                 const _Float16* __restrict__ Wb,
                                                 float* __restrict__ hf,
                                                 float* __restrict__ hb) {
  const int L = 256;
  const int dir = blockIdx.x;
  const float* xpre = dir ? xpb : xpf;
  const _Float16* Whh = dir ? Wb : Wf;
  float* hout = dir ? hb : hf;

  int tid = threadIdx.x, wave = tid >> 5, lane = tid & 31;
  int u0 = wave * 16;
  int kg = lane >> 4, nn = lane & 15;
  int u = u0 + nn;

  __shared__ __align__(16) _Float16 hs[32 * 256];
  for (int i = tid; i < 32 * 256; i += 512) hs[i] = (_Float16)0.0f;

  // register-resident B fragments: gates x K-tiles {0,1,2}
  v16h Bres[4][3];
#pragma unroll
  for (int g = 0; g < 4; ++g)
#pragma unroll
    for (int kt = 0; kt < 3; ++kt)
      Bres[g][kt] = load_frag(Whh, 256, g * 256 + u0, kt * 32);

  v8f cst[2];
  cst[0] = vzero8(); cst[1] = vzero8();
  __syncthreads();

  for (int s = 0; s < L; ++s) {
    int t = dir ? (L - 1 - s) : s;
    asm volatile("" ::: "memory");  // keep streamed W_hh loads inside the loop

    // prefetch this step's x_pre gate rows for our lanes (hides strided f32 latency)
#pragma unroll
    for (int m = 0; m < 2; ++m) {
      size_t xrow = ((size_t)(m * 16 + 8 * kg) * L + t) * 1024;
#pragma unroll
      for (int g = 0; g < 4; ++g) __builtin_prefetch(xpre + xrow + g * 256 + u, 0, 1);
    }

    v8f acc[2][4];
#pragma unroll
    for (int m = 0; m < 2; ++m)
#pragma unroll
      for (int g = 0; g < 4; ++g) acc[m][g] = vzero8();

    // phase 1: register-resident K-tiles (no memory traffic for B)
#pragma unroll
    for (int kt = 0; kt < 3; ++kt) {
      v16h a0 = load_frag(hs, 256, 0, kt * 32);    // ds_load_b128 x2
      v16h a1 = load_frag(hs, 256, 16, kt * 32);
#pragma unroll
      for (int g = 0; g < 4; ++g) {
        acc[0][g] = WMMA_F16(a0, Bres[g][kt], acc[0][g]);
        acc[1][g] = WMMA_F16(a1, Bres[g][kt], acc[1][g]);
      }
    }
    // phase 2: streamed K-tiles (L2-resident W_hh), rolled to bound VGPR pressure
#pragma unroll 1
    for (int kt = 3; kt < 8; ++kt) {
      v16h a0 = load_frag(hs, 256, 0, kt * 32);
      v16h a1 = load_frag(hs, 256, 16, kt * 32);
#pragma unroll
      for (int g = 0; g < 4; ++g) {
        v16h b = load_frag(Whh, 256, g * 256 + u0, kt * 32);
        acc[0][g] = WMMA_F16(a0, b, acc[0][g]);
        acc[1][g] = WMMA_F16(a1, b, acc[1][g]);
      }
    }
    __syncthreads();                          // all reads of hs complete

#pragma unroll
    for (int m = 0; m < 2; ++m) {
#pragma unroll
      for (int r = 0; r < 8; ++r) {
        int b = m * 16 + r + 8 * kg;          // batch index 0..31
        size_t xrow = ((size_t)b * L + t) * 1024;
        float gi = acc[m][0][r] + xpre[xrow +   0 + u];
        float gf = acc[m][1][r] + xpre[xrow + 256 + u];
        float gg = acc[m][2][r] + xpre[xrow + 512 + u];
        float go = acc[m][3][r] + xpre[xrow + 768 + u];
        float iv = sigm(gi), fv = sigm(gf), gv = tanhf_(gg), ov = sigm(go);
        float c = fv * cst[m][r] + iv * gv;
        cst[m][r] = c;
        float h = ov * tanhf_(c);
        hs[b * 256 + u] = (_Float16)h;
        hout[((size_t)b * L + t) * 256 + u] = h;
      }
    }
    __syncthreads();                          // hs updated before next step
  }
}

// ---------------- emissions: em[l,b,j] = [hf|hb] . emit_W[j] + emit_b[j] ----------------
__global__ void emit_kernel(const float* __restrict__ hf, const float* __restrict__ hb,
                            const float* __restrict__ eW, const float* __restrict__ eb,
                            float* __restrict__ em, int L) {
  int gid = blockIdx.x * blockDim.x + threadIdx.x;
  if (gid >= L * 32 * 9) return;
  int j = gid % 9, b = (gid / 9) % 32, l = gid / (9 * 32);
  size_t hrow = ((size_t)b * L + l) * 256;
  const float* wr = eW + (size_t)j * 512;
  float s = eb[j];
  for (int u = 0; u < 256; ++u) s += hf[hrow + u] * wr[u];
  for (int u = 0; u < 256; ++u) s += hb[hrow + u] * wr[256 + u];
  em[((size_t)l * 32 + b) * 9 + j] = s;
}

// ---------------- CRF NLL: single block, thread = (batch b, tag j) ----------------
__global__ __launch_bounds__(288) void crf_kernel(const float* __restrict__ em,
                                                  const int* __restrict__ mask,
                                                  const int* __restrict__ tags,
                                                  const float* __restrict__ start,
                                                  const float* __restrict__ endv,
                                                  const float* __restrict__ trans,
                                                  float* __restrict__ out, int L) {
  int tid = threadIdx.x;
  int b = tid / 9, j = tid % 9;
  __shared__ float sc[32][9];
  __shared__ float tr[81];
  __shared__ float den_s[32], num_s[32];
  for (int i = tid; i < 81; i += 288) tr[i] = trans[i];
  __syncthreads();

  float s = start[j] + em[(size_t)b * 9 + j];
  for (int t = 1; t < L; ++t) {
    sc[b][j] = s;
    __syncthreads();
    float mx = -1e30f;
#pragma unroll
    for (int i = 0; i < 9; ++i) mx = fmaxf(mx, sc[b][i] + tr[i * 9 + j]);
    float sm = 0.0f;
#pragma unroll
    for (int i = 0; i < 9; ++i) sm += __expf(sc[b][i] + tr[i * 9 + j] - mx);
    float nxt = mx + __logf(sm) + em[((size_t)t * 32 + b) * 9 + j];
    if (mask[b * L + t] > 0) s = nxt;
    __syncthreads();
  }
  sc[b][j] = s + endv[j];
  __syncthreads();

  if (j == 0) {
    float mx = -1e30f;
#pragma unroll
    for (int i = 0; i < 9; ++i) mx = fmaxf(mx, sc[b][i]);
    float sm = 0.0f;
#pragma unroll
    for (int i = 0; i < 9; ++i) sm += __expf(sc[b][i] - mx);
    den_s[b] = mx + __logf(sm);

    // gold path score
    int prev = tags[b * L];
    float num = start[prev] + em[(size_t)b * 9 + prev];
    int msum = mask[b * L];
    for (int t = 1; t < L; ++t) {
      int cur = tags[b * L + t];
      float mf = (float)mask[b * L + t];
      num += mf * (tr[prev * 9 + cur] + em[((size_t)t * 32 + b) * 9 + cur]);
      prev = cur;
      msum += mask[b * L + t];
    }
    int send = msum - 1;
    num += endv[tags[b * L + send]];
    num_s[b] = num;
  }
  __syncthreads();
  if (tid == 0) {
    float tot = 0.0f;
    for (int i = 0; i < 32; ++i) tot += den_s[i] - num_s[i];
    out[0] = tot;
  }
}

// ---------------- host launcher ----------------
extern "C" void kernel_launch(void* const* d_in, const int* in_sizes, int n_in,
                              void* d_out, int out_size, void* d_ws, size_t ws_size,
                              hipStream_t stream) {
  const int B = 32, L = 256, E = 400, Ep = 448, H = 256, G4 = 1024, T = 9;
  const int M = B * L;  // 8192

  const int*   wids   = (const int*)  d_in[0];
  const int*   gids   = (const int*)  d_in[1];
  const int*   masks  = (const int*)  d_in[2];
  const int*   labels = (const int*)  d_in[3];
  const float* w2v    = (const float*)d_in[4];
  const float* glove  = (const float*)d_in[5];
  const float* denseW = (const float*)d_in[6];
  const float* denseb = (const float*)d_in[7];
  const float* Wihf   = (const float*)d_in[8];
  const float* Whhf   = (const float*)d_in[9];
  const float* bf     = (const float*)d_in[10];
  const float* Wihb   = (const float*)d_in[11];
  const float* Whhb   = (const float*)d_in[12];
  const float* bb     = (const float*)d_in[13];
  const float* emitW  = (const float*)d_in[14];
  const float* emitb  = (const float*)d_in[15];
  const float* cstart = (const float*)d_in[16];
  const float* cend   = (const float*)d_in[17];
  const float* ctrans = (const float*)d_in[18];
  float* out = (float*)d_out;

  size_t off = 0;
  auto alloc = [&](size_t bytes) -> void* {
    void* p = (char*)d_ws + off;
    off += (bytes + 255) & ~(size_t)255;
    return p;
  };
  _Float16* emb16   = (_Float16*)alloc((size_t)M * Ep * 2);
  _Float16* x16     = (_Float16*)alloc((size_t)M * Ep * 2);
  _Float16* dW16    = (_Float16*)alloc((size_t)Ep * Ep * 2);   // N padded to 448 too
  float*    db_pad  = (float*)   alloc((size_t)Ep * 4);
  _Float16* Wihf16  = (_Float16*)alloc((size_t)G4 * Ep * 2);
  _Float16* Wihb16  = (_Float16*)alloc((size_t)G4 * Ep * 2);
  _Float16* Whhf16  = (_Float16*)alloc((size_t)G4 * H * 2);
  _Float16* Whhb16  = (_Float16*)alloc((size_t)G4 * H * 2);
  float*    xpf     = (float*)   alloc((size_t)M * G4 * 4);
  float*    xpb     = (float*)   alloc((size_t)M * G4 * 4);
  float*    hfbuf   = (float*)   alloc((size_t)M * H * 4);
  float*    hbbuf   = (float*)   alloc((size_t)M * H * 4);
  float*    embuf   = (float*)   alloc((size_t)M * T * 4);
  (void)ws_size; (void)n_in; (void)in_sizes; (void)out_size;

  auto blks = [](long n, int bs) { return (int)((n + bs - 1) / bs); };

  // 1) weight conversion / padding (all GEMM dims become exactly tile-divisible)
  cvt_pad_f16<<<blks((long)Ep * Ep, 256), 256, 0, stream>>>(denseW, dW16, E, E, Ep, Ep);
  cvt_pad_f16<<<blks((long)G4 * Ep, 256), 256, 0, stream>>>(Wihf, Wihf16, G4, E, G4, Ep);
  cvt_pad_f16<<<blks((long)G4 * Ep, 256), 256, 0, stream>>>(Wihb, Wihb16, G4, E, G4, Ep);
  cvt_pad_f16<<<blks((long)G4 * H, 256), 256, 0, stream>>>(Whhf, Whhf16, G4, H, G4, H);
  cvt_pad_f16<<<blks((long)G4 * H, 256), 256, 0, stream>>>(Whhb, Whhb16, G4, H, G4, H);
  pad_bias<<<blks(Ep, 256), 256, 0, stream>>>(denseb, db_pad, E, Ep);

  // 2) embedding gather -> f16 (cols padded to 448)
  embed_gather<<<blks((long)M * Ep, 256), 256, 0, stream>>>(wids, gids, w2v, glove, emb16, M, Ep);

  // 3) dense projection: x16 = emb16 @ dW16^T + db  (f16 out, zero-padded cols)
  gemm_nt<false, true><<<dim3(M / 128, Ep / 64), 256, 0, stream>>>(
      emb16, Ep, dW16, Ep, db_pad, nullptr, x16, Ep, Ep);

  // 4) input-to-hidden pre-activations (f32)
  gemm_nt<true, false><<<dim3(M / 128, G4 / 64), 256, 0, stream>>>(
      x16, Ep, Wihf16, Ep, bf, xpf, nullptr, G4, Ep);
  gemm_nt<true, false><<<dim3(M / 128, G4 / 64), 256, 0, stream>>>(
      x16, Ep, Wihb16, Ep, bb, xpb, nullptr, G4, Ep);

  // 5) both LSTM directions concurrently (one workgroup each)
  lstm_scan<<<dim3(2), 512, 0, stream>>>(xpf, xpb, Whhf16, Whhb16, hfbuf, hbbuf);

  // 6) emissions
  emit_kernel<<<blks((long)M * T, 256), 256, 0, stream>>>(hfbuf, hbbuf, emitW, emitb, embuf, L);

  // 7) CRF negative log-likelihood -> scalar
  crf_kernel<<<1, 288, 0, stream>>>(embuf, masks, labels, cstart, cend, ctrans, out, L);
}
